// LSTMEncoder2_26603027431884
// MI455X (gfx1250) — matrix operand order repaired and checked by reading, not compiled, not run on hardware
//
#include <hip/hip_runtime.h>
#include <hip/hip_bf16.h>

typedef __attribute__((ext_vector_type(16))) _Float16 v16h;
typedef __attribute__((ext_vector_type(4)))  _Float16 v4h;
typedef __attribute__((ext_vector_type(2)))  _Float16 v2h;
typedef __attribute__((ext_vector_type(8)))  float    v8f;

#define BB  512
#define TT  512
#define FF  32
#define HH  128
#define EE  64
#define NG1 (4*HH)   // 512 gate cols layer 1
#define NG2 (4*EE)   // 256 gate cols layer 2

#define KT_IH1 1            // F/32
#define NT_IH1 (NG1/16)     // 32
#define KT_HH1 (HH/32)      // 4
#define KT_IH2 (HH/32)      // 4
#define NT_IH2 (NG2/16)     // 16
#define KT_HH2 (EE/32)      // 2

#define NTHREADS 512        // 16 wave32s -> 4 waves/SIMD on the WGP

__device__ __forceinline__ float tanh_fast(float x) {
#if __has_builtin(__builtin_amdgcn_tanhf)
    return __builtin_amdgcn_tanhf(x);           // gfx1250 V_TANH_F32 (1 trans op)
#else
    float e = __expf(-2.0f * x);
    return (1.0f - e) * __builtin_amdgcn_rcpf(1.0f + e);
#endif
}
__device__ __forceinline__ float sigm(float x) {
    return 0.5f + 0.5f * tanh_fast(0.5f * x);   // 1 trans op
}

// B-fragment (weight) packed layout: frag (nt,kt) -> 32 lanes x 16 halves.
// lane's element e maps to B[k][n] = W[n][k] with
//   n = nt*16 + (lane&15),  k = kt*32 + 16*(lane>>4) + e
__device__ void pack_weight(const float* __restrict__ W, _Float16* dst,
                            int Kdim, int NT, int KT, int tid, int nthreads) {
    int total = NT * KT * 32;
    for (int i = tid; i < total; i += nthreads) {
        int lane = i & 31;
        int frag = i >> 5;
        int nt = frag / KT, kt = frag - nt * KT;
        int n  = nt * 16 + (lane & 15);
        int kb = kt * 32 + ((lane >> 4) << 4);
        const float* src = W + (size_t)n * Kdim + kb;
        _Float16* d = dst + (size_t)i * 16;
#pragma unroll
        for (int e = 0; e < 16; ++e) d[e] = (_Float16)src[e];
    }
}

__device__ __forceinline__ v16h load_b_frag(const _Float16* w, int nt, int kt, int KT, int lane) {
    const _Float16* p = w + (((nt * KT + kt) * 32 + lane) << 4);
    return *(const v16h*)p;
}

// A fragment (16-bit, 16x32): lanes 0-15 row=lane, K = {0..15};
// lanes 16-31 row=lane-16, K = {16..31}
__device__ __forceinline__ v16h load_a_frag(const _Float16* m, int stride, int kt, int lane) {
    int row = lane & 15;
    int kb  = kt * 32 + ((lane >> 4) << 4);
    v16h a;
    *(uint4*)&a       = *(const uint4*)(m + row * stride + kb);
    *(((uint4*)&a)+1) = *(const uint4*)(m + row * stride + kb + 8);
    return a;
}

__device__ __forceinline__ v8f wmma_f16(v16h a, v16h b, v8f c) {
    return __builtin_amdgcn_wmma_f32_16x16x32_f16(false, a, false, b, (short)0, c, false, false);
}

__global__ __launch_bounds__(NTHREADS, 1)
void lstm2_fused_kernel(const float* __restrict__ x,
                        const float* __restrict__ Wih1, const float* __restrict__ Whh1,
                        const float* __restrict__ bih1, const float* __restrict__ bhh1,
                        const float* __restrict__ Wih2, const float* __restrict__ Whh2,
                        const float* __restrict__ bih2, const float* __restrict__ bhh2,
                        float* __restrict__ out)
{
    // ---- LDS: ~315 KB total (fits CDNA5 320KB/WGP) ----
    __shared__ __align__(32) _Float16 sWih1[NT_IH1 * KT_IH1 * 512]; //  32 KB
    __shared__ __align__(32) _Float16 sWhh1[NT_IH1 * KT_HH1 * 512]; // 128 KB
    __shared__ __align__(32) _Float16 sWih2[NT_IH2 * KT_IH2 * 512]; //  64 KB
    __shared__ __align__(32) _Float16 sWhh2[NT_IH2 * KT_HH2 * 512]; //  32 KB
    __shared__ __align__(16) float    sB1[NG1];                     //   2 KB
    __shared__ __align__(16) float    sB2[NG2];                     //   1 KB
    __shared__ __align__(32) _Float16 sX [2][16 * FF];              //   2 KB (double buffer)
    __shared__ __align__(32) _Float16 sH1[16 * HH];                 //   4 KB
    __shared__ __align__(32) _Float16 sH2[16 * EE];                 //   2 KB
    __shared__ __align__(16) float    sG1[16 * NG1];                //  32 KB
    __shared__ __align__(16) float    sG2[16 * NG2];                //  16 KB

    const int tid    = threadIdx.x;
    const int lane   = tid & 31;
    const int wave   = tid >> 5;          // 0..15
    const int batch0 = blockIdx.x * 16;

    // ---- prologue: pack weights to f16 fragment layout, sum biases, zero h ----
    pack_weight(Wih1, sWih1, FF, NT_IH1, KT_IH1, tid, NTHREADS);
    pack_weight(Whh1, sWhh1, HH, NT_IH1, KT_HH1, tid, NTHREADS);
    pack_weight(Wih2, sWih2, HH, NT_IH2, KT_IH2, tid, NTHREADS);
    pack_weight(Whh2, sWhh2, EE, NT_IH2, KT_HH2, tid, NTHREADS);
    for (int j = tid; j < NG1; j += NTHREADS) sB1[j] = bih1[j] + bhh1[j];
    for (int j = tid; j < NG2; j += NTHREADS) sB2[j] = bih2[j] + bhh2[j];
    for (int j = tid; j < 16 * HH; j += NTHREADS) sH1[j] = (_Float16)0.0f;
    for (int j = tid; j < 16 * EE; j += NTHREADS) sH2[j] = (_Float16)0.0f;

    // per-thread cell state (owned elementwise slots)
    float c1[4], c2[2], h2r[2];
#pragma unroll
    for (int i = 0; i < 4; ++i) c1[i] = 0.0f;
#pragma unroll
    for (int i = 0; i < 2; ++i) { c2[i] = 0.0f; h2r[i] = 0.0f; }

    // elementwise ownership: L1 -> 4 contiguous cols/thread, L2 -> 2 cols/thread
    const int r1  = tid >> 5;             // row 0..15 (one row per wave)
    const int cc1 = lane * 4;             // cols within H=128
    const int r2  = tid >> 5;
    const int cc2 = lane * 2;             // cols within E=64

    // x staging ownership (1 element/thread) + register prefetch pipeline
    const int xr = tid >> 5, xc = lane;
    const float* xrow = x + ((size_t)(batch0 + xr)) * TT * FF + xc;
    float xnext = xrow[0];                // prefetch t = 0

    __syncthreads();

    // persistent bias fragments in VGPRs (C operand of first WMMA in each chain)
    v8f b1f[2], b2f;
#pragma unroll
    for (int i = 0; i < 2; ++i) {
        float bv = sB1[(wave * 2 + i) * 16 + (lane & 15)];
        b1f[i] = (v8f){bv, bv, bv, bv, bv, bv, bv, bv};
    }
    {
        float bv = sB2[wave * 16 + (lane & 15)];
        b2f = (v8f){bv, bv, bv, bv, bv, bv, bv, bv};
    }

    int buf = 0;
    for (int t = 0; t < TT; ++t) {
        // ---- stage prefetched x[t], then issue prefetch of x[t+1] (hidden by step body) ----
        sX[buf][xr * FF + xc] = (_Float16)xnext;
        if (t + 1 < TT) xnext = xrow[(size_t)(t + 1) * FF];
        __syncthreads();                                   // A

        // ---- layer-1 gates: wave handles cols [32w, 32w+32) ----
#pragma unroll
        for (int i = 0; i < 2; ++i) {
            int nt   = wave * 2 + i;
            int ncol = nt * 16 + (lane & 15);
            // x contribution (K = 32), bias as C
            v8f acc = wmma_f16(load_a_frag(sX[buf], FF, 0, lane),
                               load_b_frag(sWih1, nt, 0, KT_IH1, lane), b1f[i]);
            // recurrent contribution (K = 128)
#pragma unroll
            for (int kt = 0; kt < KT_HH1; ++kt)
                acc = wmma_f16(load_a_frag(sH1, HH, kt, lane),
                               load_b_frag(sWhh1, nt, kt, KT_HH1, lane), acc);
            int mbase = (lane >> 4) << 3;
#pragma unroll
            for (int v = 0; v < 8; ++v)
                sG1[(mbase + v) * NG1 + ncol] = acc[v];
        }

        // ---- layer-2 RECURRENT contribution computed early (prev-step h2 is stable
        //      from barrier A until elementwise-2); result carried in registers ----
        v8f acc2 = wmma_f16(load_a_frag(sH2, EE, 0, lane),
                            load_b_frag(sWhh2, wave, 0, KT_HH2, lane), b2f);
        acc2     = wmma_f16(load_a_frag(sH2, EE, 1, lane),
                            load_b_frag(sWhh2, wave, 1, KT_HH2, lane), acc2);
        __syncthreads();                                   // B

        // ---- layer-1 elementwise (i,f,g,o) + h1 -> LDS f16 (vectorized LDS access) ----
        {
            const float* g = sG1 + r1 * NG1;
            float4 gi = *(const float4*)(g + cc1);
            float4 gf = *(const float4*)(g + HH     + cc1);
            float4 gg = *(const float4*)(g + 2 * HH + cc1);
            float4 go = *(const float4*)(g + 3 * HH + cc1);
            float iv[4] = {gi.x, gi.y, gi.z, gi.w};
            float fv[4] = {gf.x, gf.y, gf.z, gf.w};
            float gv[4] = {gg.x, gg.y, gg.z, gg.w};
            float ov[4] = {go.x, go.y, go.z, go.w};
            v4h hv;
#pragma unroll
            for (int j = 0; j < 4; ++j) {
                float c = sigm(fv[j]) * c1[j] + sigm(iv[j]) * tanh_fast(gv[j]);
                c1[j] = c;
                hv[j] = (_Float16)(sigm(ov[j]) * tanh_fast(c));
            }
            *(v4h*)(&sH1[r1 * HH + cc1]) = hv;             // single 64-bit store
        }
        __syncthreads();                                   // C

        // ---- layer-2 gates: input contribution (K = 128) chained onto early acc2 ----
        {
            int ncol = wave * 16 + (lane & 15);
#pragma unroll
            for (int kt = 0; kt < KT_IH2; ++kt)
                acc2 = wmma_f16(load_a_frag(sH1, HH, kt, lane),
                                load_b_frag(sWih2, wave, kt, KT_IH2, lane), acc2);
            int mbase = (lane >> 4) << 3;
#pragma unroll
            for (int v = 0; v < 8; ++v)
                sG2[(mbase + v) * NG2 + ncol] = acc2[v];
        }
        __syncthreads();                                   // D

        // ---- layer-2 elementwise + h2 -> LDS f16, keep f32 copy for output ----
        {
            const float* g = sG2 + r2 * NG2;
            float2 gi = *(const float2*)(g + cc2);
            float2 gf = *(const float2*)(g + EE     + cc2);
            float2 gg = *(const float2*)(g + 2 * EE + cc2);
            float2 go = *(const float2*)(g + 3 * EE + cc2);
            float iv[2] = {gi.x, gi.y};
            float fv[2] = {gf.x, gf.y};
            float gv[2] = {gg.x, gg.y};
            float ov[2] = {go.x, go.y};
            v2h hv;
#pragma unroll
            for (int j = 0; j < 2; ++j) {
                float c = sigm(fv[j]) * c2[j] + sigm(iv[j]) * tanh_fast(gv[j]);
                c2[j] = c;
                float h = sigm(ov[j]) * tanh_fast(c);
                h2r[j] = h;
                hv[j] = (_Float16)h;
            }
            *(v2h*)(&sH2[r2 * EE + cc2]) = hv;             // single 32-bit store
        }
        buf ^= 1;   // next step stages into the other x buffer; no end barrier needed
    }

    // ---- write final hidden state of layer 2: [B, E] f32 ----
#pragma unroll
    for (int j = 0; j < 2; ++j)
        out[(size_t)(batch0 + r2) * EE + cc2 + j] = h2r[j];
}

extern "C" void kernel_launch(void* const* d_in, const int* in_sizes, int n_in,
                              void* d_out, int out_size, void* d_ws, size_t ws_size,
                              hipStream_t stream) {
    const float* x    = (const float*)d_in[0];
    const float* Wih1 = (const float*)d_in[1];
    const float* Whh1 = (const float*)d_in[2];
    const float* bih1 = (const float*)d_in[3];
    const float* bhh1 = (const float*)d_in[4];
    const float* Wih2 = (const float*)d_in[5];
    const float* Whh2 = (const float*)d_in[6];
    const float* bih2 = (const float*)d_in[7];
    const float* bhh2 = (const float*)d_in[8];
    float* out = (float*)d_out;

    hipLaunchKernelGGL(lstm2_fused_kernel, dim3(BB / 16), dim3(NTHREADS), 0, stream,
                       x, Wih1, Whh1, bih1, bhh1, Wih2, Whh2, bih2, bhh2, out);
}
